// AttentionSuper_8074538516486
// MI455X (gfx1250) — matrix operand order repaired and checked by reading, not compiled, not run on hardware
//
#include <hip/hip_runtime.h>
#include <hip/hip_bf16.h>

#define BB 64
#define NQ 197
#define CC 640
#define HH 10
#define HD 64
#define NP 208          // 197 padded to 13*16
#define TT 30
#define SCALE_F 0.125f  // 64^-0.5
#define MROWS (BB * NQ) // 12608 = 788*16 = 197*64 exactly
#define KSTEPS (CC / 32)

typedef __attribute__((ext_vector_type(16))) _Float16 v16h;
typedef __attribute__((ext_vector_type(8)))  _Float16 v8h;
typedef __attribute__((ext_vector_type(8)))  float    v8f;
typedef __attribute__((__vector_size__(16))) int      i32x4;

#if defined(__has_builtin)
#if __has_builtin(__builtin_amdgcn_global_load_async_to_lds_b128)
#define HAS_ASYNC_LDS 1
#endif
#endif

// ---------------------------------------------------------------------------
// WMMA wrapper: D = A(16x32 f16) * B(32x16 f16) + C(16x16 f32)
static __device__ __forceinline__ v8f wmma16(v16h a, v16h b, v8f c) {
    return __builtin_amdgcn_wmma_f32_16x16x32_f16(false, a, false, b,
                                                  (short)0, c, false, false);
}

// A-fragment: lane L holds row M = L%16; halves map to
// K = 16*hi + 0..15 (hi = L>=16) -> two 16B loads.
static __device__ __forceinline__ v16h ld_a16(const _Float16* row, int k0, int hi) {
    const _Float16* p = row + k0 + hi * 16;
    v8h lo = *(const v8h*)(p);
    v8h hn = *(const v8h*)(p + 8);
    v16h a;
#pragma unroll
    for (int i = 0; i < 8; ++i) { a[i] = lo[i]; a[8 + i] = hn[i]; }
    return a;
}
// B-fragment: lane L holds column N = L%16; halves K = 16*hi + h.
// 16B-aligned variant (LDS rows with 40-half pitch): two v8h loads.
static __device__ __forceinline__ v16h ld_b16(const _Float16* p) {
    v8h lo = *(const v8h*)(p);
    v8h hn = *(const v8h*)(p + 8);
    v16h b;
#pragma unroll
    for (int i = 0; i < 8; ++i) { b[i] = lo[i]; b[8 + i] = hn[i]; }
    return b;
}

// Relative-position indices (mirrors _rel_indices: n=197, max_rel=14, s=14)
static __device__ __forceinline__ void rel_idx(int qq, int kk, int& iv, int& ih) {
    if (qq == 0 || kk == 0) { iv = 0; ih = 0; return; }   // CLS pad row/col
    int a = qq - 1, b = kk - 1;
    int dv = (b / 14) - (a / 14);
    int dh = (b % 14) - (a % 14);
    dv = dv < -14 ? -14 : (dv > 14 ? 14 : dv);
    dh = dh < -14 ? -14 : (dh > 14 ? 14 : dh);
    iv = dv + 15; ih = dh + 15;
}

// ---------------------------------------------------------------------------
// Elementwise helpers
__global__ void cvt_f32_f16(const float* __restrict__ s, _Float16* __restrict__ d, int n) {
    int i = blockIdx.x * blockDim.x + threadIdx.x;
    int st = gridDim.x * blockDim.x;
    for (; i < n; i += st) d[i] = (_Float16)s[i];
}
__global__ void zero_f16(_Float16* __restrict__ d, int n) {
    int i = blockIdx.x * blockDim.x + threadIdx.x;
    int st = gridDim.x * blockDim.x;
    for (; i < n; i += st) d[i] = (_Float16)0.f;
}
__global__ void pack_relk(const float* __restrict__ s, _Float16* __restrict__ d) {
    int i = blockIdx.x * blockDim.x + threadIdx.x;
    if (i >= 32 * 64) return;
    int t = i >> 6, dd = i & 63;
    d[i] = (t < TT) ? (_Float16)s[t * HD + dd] : (_Float16)0.f;
}
__global__ void pack_relvT(const float* __restrict__ s, _Float16* __restrict__ d) {
    int i = blockIdx.x * blockDim.x + threadIdx.x;
    if (i >= 64 * 32) return;
    int dd = i >> 5, t = i & 31;
    d[i] = (t < TT) ? (_Float16)s[t * HD + dd] : (_Float16)0.f;
}

// ---------------------------------------------------------------------------
// 64x64-tile GEMM core, 4-wave workgroup, LDS-staged 64x32 panels (pitch 40
// halves for conflict-free 16B fragment reads). Async LDS path when available.
#if defined(HAS_ASYNC_LDS)
static __device__ __forceinline__ void async_b128(const _Float16* g, _Float16* l) {
    __builtin_amdgcn_global_load_async_to_lds_b128(
        (__attribute__((address_space(1))) i32x4*)(g),
        (__attribute__((address_space(3))) i32x4*)(l),
        0, 0);
}
static __device__ __forceinline__ void stage_async(
    const _Float16* __restrict__ Ab, const _Float16* __restrict__ Bb, int k0,
    _Float16 (*As)[40], _Float16 (*Bs)[40], int tid)
{
#pragma unroll
    for (int cix = 0; cix < 2; ++cix) {
        int idx = tid + 128 * cix;            // 0..255 -> 64 rows x 4 16B segs
        int row = idx >> 2, seg = (idx & 3) * 8;
        async_b128(Ab + (size_t)row * CC + k0 + seg, &As[row][seg]);
        async_b128(Bb + (size_t)row * CC + k0 + seg, &Bs[row][seg]);
    }
}
#endif

static __device__ __forceinline__ void gemm64_core(
    const _Float16* __restrict__ Ab, const _Float16* __restrict__ Bb,
    _Float16 (*As)[64][40], _Float16 (*Bs)[64][40], int tid, v8f cc[4])
{
    const int lane = tid & 31, w = tid >> 5;
    const int hi = lane >> 4, n = lane & 15;
#if defined(HAS_ASYNC_LDS)
    stage_async(Ab, Bb, 0, As[0], Bs[0], tid);
    for (int s = 0; s < KSTEPS; ++s) {
        const int buf = s & 1;
        if (s + 1 < KSTEPS) {
            stage_async(Ab, Bb, 32 * (s + 1), As[buf ^ 1], Bs[buf ^ 1], tid);
#if __has_builtin(__builtin_amdgcn_s_wait_asynccnt)
            __builtin_amdgcn_s_wait_asynccnt(4);
#else
            asm volatile("s_wait_asynccnt 0x4" ::: "memory");
#endif
        } else {
#if __has_builtin(__builtin_amdgcn_s_wait_asynccnt)
            __builtin_amdgcn_s_wait_asynccnt(0);
#else
            asm volatile("s_wait_asynccnt 0x0" ::: "memory");
#endif
        }
        __syncthreads();
        v16h af = ld_a16(&As[buf][16 * w + n][0], 0, hi);
#pragma unroll
        for (int t = 0; t < 4; ++t)
            cc[t] = wmma16(af, ld_b16(&Bs[buf][16 * t + n][hi * 16]), cc[t]);
        __syncthreads();
    }
#else
    // fallback: register-pipelined global->VGPR->ds_store staging, 1 buffer
    const int row = tid >> 1, seg = (tid & 1) * 16;  // 2x b128 per thread/tile
    uint4 ra0, ra1, rb0, rb1;
    {
        const _Float16* ga = Ab + (size_t)row * CC + seg;
        const _Float16* gb = Bb + (size_t)row * CC + seg;
        ra0 = *(const uint4*)(ga);     ra1 = *(const uint4*)(ga + 8);
        rb0 = *(const uint4*)(gb);     rb1 = *(const uint4*)(gb + 8);
    }
    for (int s = 0; s < KSTEPS; ++s) {
        *(uint4*)&As[0][row][seg]     = ra0;  *(uint4*)&As[0][row][seg + 8] = ra1;
        *(uint4*)&Bs[0][row][seg]     = rb0;  *(uint4*)&Bs[0][row][seg + 8] = rb1;
        if (s + 1 < KSTEPS) {
            const _Float16* ga = Ab + (size_t)row * CC + 32 * (s + 1) + seg;
            const _Float16* gb = Bb + (size_t)row * CC + 32 * (s + 1) + seg;
            ra0 = *(const uint4*)(ga); ra1 = *(const uint4*)(ga + 8);
            rb0 = *(const uint4*)(gb); rb1 = *(const uint4*)(gb + 8);
        }
        __syncthreads();
        v16h af = ld_a16(&As[0][16 * w + n][0], 0, hi);
#pragma unroll
        for (int t = 0; t < 4; ++t)
            cc[t] = wmma16(af, ld_b16(&Bs[0][16 * t + n][hi * 16]), cc[t]);
        __syncthreads();
    }
#endif
}

// ---------------------------------------------------------------------------
// QKV GEMM: Y[m,o] = sum_c xh[m,c] * Wqkv[o,c] + b[o], scattered into
// q (pre-scaled, [B,H,NP,HD]), k ([B,H,NP,HD]), v transposed ([B,H,HD,NP]).
__global__ __launch_bounds__(128) void qkv_gemm_kernel(
    const _Float16* __restrict__ xh, const _Float16* __restrict__ wh,
    const float* __restrict__ qkv_b,
    _Float16* __restrict__ qs, _Float16* __restrict__ kh, _Float16* __restrict__ vt)
{
    __shared__ __align__(16) _Float16 As[2][64][40];
    __shared__ __align__(16) _Float16 Bs[2][64][40];
    const int o0 = blockIdx.x * 64, m0 = blockIdx.y * 64;
    const int tid = threadIdx.x, lane = tid & 31, w = tid >> 5;
    const int hi = lane >> 4, n = lane & 15;
    v8f cc[4] = {v8f{}, v8f{}, v8f{}, v8f{}};
    gemm64_core(xh + (size_t)m0 * CC, wh + (size_t)o0 * CC, As, Bs, tid, cc);

    const int j3 = o0 / CC;                  // 0=q 1=k 2=v (uniform per block)
    const int h  = (o0 - j3 * CC) >> 6;      // uniform per block
#pragma unroll
    for (int t = 0; t < 4; ++t) {
        const int d = 16 * t + n;            // o0 % 64 == 0 -> d = o % 64
        const float bias = qkv_b[o0 + d];
#pragma unroll
        for (int i = 0; i < 8; ++i) {
            int m  = m0 + 16 * w + i + hi * 8;
            int b_ = m / NQ;
            int nq = m - b_ * NQ;
            float val = cc[t][i] + bias;
            int bh = b_ * HH + h;
            if (j3 == 0)      qs[((size_t)bh * NP + nq) * HD + d] = (_Float16)(val * SCALE_F);
            else if (j3 == 1) kh[((size_t)bh * NP + nq) * HD + d] = (_Float16)val;
            else              vt[((size_t)bh * HD + d) * NP + nq] = (_Float16)val;
        }
    }
}

// Projection GEMM: out[m,o] = sum_c att[m,c] * Wp[o,c] + pb[o]  (fp32 out)
__global__ __launch_bounds__(128) void proj_gemm_kernel(
    const _Float16* __restrict__ ah, const _Float16* __restrict__ wh,
    const float* __restrict__ pb, float* __restrict__ out)
{
    __shared__ __align__(16) _Float16 As[2][64][40];
    __shared__ __align__(16) _Float16 Bs[2][64][40];
    const int o0 = blockIdx.x * 64, m0 = blockIdx.y * 64;
    const int tid = threadIdx.x, lane = tid & 31, w = tid >> 5;
    const int hi = lane >> 4, n = lane & 15;
    v8f cc[4] = {v8f{}, v8f{}, v8f{}, v8f{}};
    gemm64_core(ah + (size_t)m0 * CC, wh + (size_t)o0 * CC, As, Bs, tid, cc);
#pragma unroll
    for (int t = 0; t < 4; ++t) {
        const float bias = pb[o0 + 16 * t + n];
#pragma unroll
        for (int i = 0; i < 8; ++i)
            out[(size_t)(m0 + 16 * w + i + hi * 8) * CC + o0 + 16 * t + n] =
                cc[t][i] + bias;
    }
}

// ---------------------------------------------------------------------------
// Fused attention: one wave per (b, h, 16-row q tile).
__global__ __launch_bounds__(32) void attn_kernel(
    const _Float16* __restrict__ qs, const _Float16* __restrict__ kh,
    const _Float16* __restrict__ vt,
    const _Float16* __restrict__ rkv, const _Float16* __restrict__ rkh,
    const _Float16* __restrict__ rvvT, const _Float16* __restrict__ rvhT,
    _Float16* __restrict__ att)
{
    __shared__ __align__(16) float    qdv[16][32];
    __shared__ __align__(16) float    qdh[16][32];
    __shared__ __align__(32) _Float16 p_lds[16][224];   // 208 cols + pad
    __shared__ __align__(16) float    wv[16][32];
    __shared__ __align__(16) float    wh_[16][32];

    const int b = blockIdx.z, h = blockIdx.y, q0 = blockIdx.x * 16;
    const int lane = threadIdx.x, hi = lane >> 4, n = lane & 15;
    const int bh = b * HH + h;
    const _Float16* qbase = qs + (size_t)bh * NP * HD;
    const _Float16* kbase = kh + (size_t)bh * NP * HD;
    const _Float16* vbase = vt + (size_t)bh * HD * NP;

#pragma unroll
    for (int i = 0; i < 16; ++i) {
        (&wv[0][0])[lane + 32 * i]  = 0.f;
        (&wh_[0][0])[lane + 32 * i] = 0.f;
    }

    const _Float16* qrow = qbase + (size_t)(q0 + n) * HD;
    v16h aq0 = ld_a16(qrow, 0, hi);
    v16h aq1 = ld_a16(qrow, 32, hi);

    // dot tables: qdv[r][t] = q_r . rel_k_v[t] (tables padded 32x64)
    {
        v8f c0 = {}, c1 = {}, c2 = {}, c3 = {};
        c0 = wmma16(aq0, *(const v16h*)(rkv + (0  + n) * HD + 0  + hi * 16), c0);
        c0 = wmma16(aq1, *(const v16h*)(rkv + (0  + n) * HD + 32 + hi * 16), c0);
        c1 = wmma16(aq0, *(const v16h*)(rkv + (16 + n) * HD + 0  + hi * 16), c1);
        c1 = wmma16(aq1, *(const v16h*)(rkv + (16 + n) * HD + 32 + hi * 16), c1);
        c2 = wmma16(aq0, *(const v16h*)(rkh + (0  + n) * HD + 0  + hi * 16), c2);
        c2 = wmma16(aq1, *(const v16h*)(rkh + (0  + n) * HD + 32 + hi * 16), c2);
        c3 = wmma16(aq0, *(const v16h*)(rkh + (16 + n) * HD + 0  + hi * 16), c3);
        c3 = wmma16(aq1, *(const v16h*)(rkh + (16 + n) * HD + 32 + hi * 16), c3);
#pragma unroll
        for (int i = 0; i < 8; ++i) {
            int M = i + hi * 8;
            qdv[M][n]      = c0[i];
            qdv[M][16 + n] = c1[i];
            qdh[M][n]      = c2[i];
            qdh[M][16 + n] = c3[i];
        }
    }
    __syncthreads();

    // scores: S = (q*SCALE) @ k^T + gathered rel-k bias; mask padding
    v8f s[13];
#pragma unroll
    for (int j = 0; j < 13; ++j) {
        const _Float16* krow = kbase + (size_t)(16 * j + n) * HD + hi * 16;
        v8f acc = {};
        acc = wmma16(aq0, *(const v16h*)(krow), acc);
        acc = wmma16(aq1, *(const v16h*)(krow + 32), acc);
        const int kk = 16 * j + n;
#pragma unroll
        for (int i = 0; i < 8; ++i) {
            int M = i + hi * 8;
            int qq = q0 + M;
            if (qq < NQ && kk < NQ) {
                int iv, ih; rel_idx(qq, kk, iv, ih);
                acc[i] += qdv[M][iv] + qdh[M][ih];
            } else {
                acc[i] = -1e30f;
            }
        }
        s[j] = acc;
    }

    // softmax across the 208-wide row; C layout: row M = i+8*hi, col = 16*j+n
#pragma unroll
    for (int i = 0; i < 8; ++i) {
        const int M = i + hi * 8;
        float mx = s[0][i];
#pragma unroll
        for (int j = 1; j < 13; ++j) mx = fmaxf(mx, s[j][i]);
#pragma unroll
        for (int off = 1; off < 16; off <<= 1) mx = fmaxf(mx, __shfl_xor(mx, off, 32));
        float sum = 0.f;
#pragma unroll
        for (int j = 0; j < 13; ++j) { float p = __expf(s[j][i] - mx); s[j][i] = p; sum += p; }
#pragma unroll
        for (int off = 1; off < 16; off <<= 1) sum += __shfl_xor(sum, off, 32);
        const float rinv = 1.0f / sum;
#pragma unroll
        for (int j = 0; j < 13; ++j) p_lds[M][16 * j + n] = (_Float16)(s[j][i] * rinv);
        p_lds[M][208 + n] = (_Float16)0.f;
    }
    __syncthreads();

    // O = P @ V (A re-fragmented from LDS, B from transposed v)
    v16h ap[7];
#pragma unroll
    for (int cix = 0; cix < 7; ++cix) ap[cix] = ld_a16(&p_lds[n][0], 32 * cix, hi);
    v8f oc[4] = {v8f{}, v8f{}, v8f{}, v8f{}};
#pragma unroll
    for (int t = 0; t < 4; ++t) {
        const _Float16* vrow = vbase + (size_t)(t * 16 + n) * NP + hi * 16;
#pragma unroll
        for (int cix = 0; cix < 7; ++cix)
            oc[t] = wmma16(ap[cix], *(const v16h*)(vrow + 32 * cix), oc[t]);
    }

    // per-row histograms of P over the 30 table indices (LDS float atomics)
    for (int rr = 0; rr < 16; ++rr) {
        const int qq = q0 + rr;
        for (int kk = lane; kk < NQ; kk += 32) {
            float p = (float)p_lds[rr][kk];
            int iv, ih; rel_idx(qq, kk, iv, ih);
            atomicAdd(&wv[rr][iv], p);
            atomicAdd(&wh_[rr][ih], p);
        }
    }
    __syncthreads();

    // O += Wv @ rel_v_v + Wh @ rel_v_h
    v16h awv, awh;
#pragma unroll
    for (int hf = 0; hf < 16; ++hf) {
        int K = hf + hi * 16;
        awv[hf] = (_Float16)wv[n][K];
        awh[hf] = (_Float16)wh_[n][K];
    }
#pragma unroll
    for (int t = 0; t < 4; ++t) {
        oc[t] = wmma16(awv, *(const v16h*)(rvvT + (size_t)(t * 16 + n) * 32 + hi * 16), oc[t]);
        oc[t] = wmma16(awh, *(const v16h*)(rvhT + (size_t)(t * 16 + n) * 32 + hi * 16), oc[t]);
    }

    // store attention output as f16 in [B*N, H*HD] layout for the proj GEMM
#pragma unroll
    for (int i = 0; i < 8; ++i) {
        const int qq = q0 + i + hi * 8;
        if (qq >= NQ) continue;
        _Float16* orow = att + ((size_t)(b * NQ + qq)) * CC + h * HD + n;
#pragma unroll
        for (int t = 0; t < 4; ++t) orow[t * 16] = (_Float16)oc[t][i];
    }
}

// ---------------------------------------------------------------------------
extern "C" void kernel_launch(void* const* d_in, const int* in_sizes, int n_in,
                              void* d_out, int out_size, void* d_ws, size_t ws_size,
                              hipStream_t stream) {
    (void)in_sizes; (void)n_in; (void)out_size; (void)ws_size;
    const float* x       = (const float*)d_in[0];
    const float* qkv_w   = (const float*)d_in[1];
    const float* qkv_b   = (const float*)d_in[2];
    const float* proj_w  = (const float*)d_in[3];
    const float* proj_b  = (const float*)d_in[4];
    const float* rel_k_v = (const float*)d_in[5];
    const float* rel_k_h = (const float*)d_in[6];
    const float* rel_v_v = (const float*)d_in[7];
    const float* rel_v_h = (const float*)d_in[8];

    char* ws = (char*)d_ws;
    size_t off = 0;
    auto carve = [&](size_t bytes) {
        char* p = ws + off;
        off += (bytes + 255) & ~(size_t)255;
        return p;
    };
    const size_t nXH  = (size_t)MROWS * CC;
    const size_t nQKV = (size_t)BB * HH * NP * HD;
    _Float16* xh   = (_Float16*)carve(nXH * 2);       // reused as attention out
    _Float16* wq   = (_Float16*)carve((size_t)3 * CC * CC * 2);
    _Float16* wp   = (_Float16*)carve((size_t)CC * CC * 2);
    _Float16* qsb  = (_Float16*)carve(nQKV * 2);
    _Float16* khb  = (_Float16*)carve(nQKV * 2);
    _Float16* vtb  = (_Float16*)carve(nQKV * 2);
    _Float16* rkv  = (_Float16*)carve(32 * 64 * 2);
    _Float16* rkh  = (_Float16*)carve(32 * 64 * 2);
    _Float16* rvvT = (_Float16*)carve(64 * 32 * 2);
    _Float16* rvhT = (_Float16*)carve(64 * 32 * 2);

    cvt_f32_f16<<<4096, 256, 0, stream>>>(x, xh, (int)nXH);
    cvt_f32_f16<<<2048, 256, 0, stream>>>(qkv_w, wq, 3 * CC * CC);
    cvt_f32_f16<<<1024, 256, 0, stream>>>(proj_w, wp, CC * CC);
    pack_relk <<<8, 256, 0, stream>>>(rel_k_v, rkv);
    pack_relk <<<8, 256, 0, stream>>>(rel_k_h, rkh);
    pack_relvT<<<8, 256, 0, stream>>>(rel_v_v, rvvT);
    pack_relvT<<<8, 256, 0, stream>>>(rel_v_h, rvhT);
    zero_f16<<<4096, 256, 0, stream>>>(qsb, (int)nQKV);
    zero_f16<<<4096, 256, 0, stream>>>(khb, (int)nQKV);
    zero_f16<<<4096, 256, 0, stream>>>(vtb, (int)nQKV);

    // QKV projection (WMMA, 64x64 tiles): 30 x 197 workgroups
    qkv_gemm_kernel<<<dim3(3 * CC / 64, MROWS / 64), 128, 0, stream>>>(
        xh, wq, qkv_b, qsb, khb, vtb);

    // fused attention (writes f16 att into the xh buffer, now dead)
    attn_kernel<<<dim3(NP / 16, HH, BB), 32, 0, stream>>>(
        qsb, khb, vtb, rkv, rkh, rvvT, rvhT, xh);

    // output projection (WMMA, 64x64 tiles) + bias -> fp32 d_out [B,N,C]
    proj_gemm_kernel<<<dim3(CC / 64, MROWS / 64), 128, 0, stream>>>(
        xh, wp, proj_b, (float*)d_out);
}
